// LSTMModel_13443247636553
// MI455X (gfx1250) — matrix operand order repaired and checked by reading, not compiled, not run on hardware
//
#include <hip/hip_runtime.h>
#include <stdint.h>

// Hidden sizes / problem dims from the reference (hard-coded there as well).
#define H1c 51
#define Tc  2048

typedef __attribute__((ext_vector_type(16))) _Float16 v16h;
typedef __attribute__((ext_vector_type(8)))  _Float16 v8h;
typedef __attribute__((ext_vector_type(2)))  _Float16 h2v;
typedef __attribute__((ext_vector_type(8)))  float    v8f;

// ---- activations: native v_tanh_f32 on gfx1250 (confirmed in codegen) ----
#if __has_builtin(__builtin_amdgcn_tanhf)
__device__ __forceinline__ float tanh1(float x) { return __builtin_amdgcn_tanhf(x); }
#else
__device__ __forceinline__ float tanh1(float x) {
  float e = __builtin_amdgcn_exp2f(2.88539008f * x);
  return 1.0f - 2.0f * __builtin_amdgcn_rcpf(e + 1.0f);
}
#endif
// sigmoid of (2*x): gate rows pre-scaled by 0.5 in the weights, so
// sig(gate) = 0.5*tanh(0.5*gate) + 0.5 = fma(tanh(x), 0.5, 0.5)
__device__ __forceinline__ float sigh(float x) {
  return __builtin_fmaf(0.5f, tanh1(x), 0.5f);
}

__device__ __forceinline__ uint32_t pkf16(float a, float b) {
  h2v p;
  p[0] = (_Float16)a;       // fused to v_cvt_pk_f16_f32
  p[1] = (_Float16)b;
  return __builtin_bit_cast(uint32_t, p);
}

__device__ __forceinline__ v16h zero16() {
  v16h z;
  #pragma unroll
  for (int e = 0; e < 16; ++e) z[e] = (_Float16)0.0f;
  return z;
}

union U16 { v16h v; uint32_t u[8]; };

// K-slot permutation: swap bits 3 and 4. Applied identically to the K axis of
// the A (weight) storage and the B (state) assembly, so the WMMA dot product
// is unchanged — but B-slot lane-half then matches the D-layout lane-half of
// the state, removing all cross-lane exchanges from the recurrence.
__device__ __forceinline__ int swap34(int k) {
  return (k & ~0x18) | ((k & 8) << 1) | ((k & 16) >> 1);
}

#define WMMA(A, B, C) \
  __builtin_amdgcn_wmma_f32_16x16x32_f16(false, (A), false, (B), (short)0, (C), false, false)

__global__ __launch_bounds__(128, 1) void lstm2_wmma_kernel(
    const float* __restrict__ input,
    const float* __restrict__ Wih1, const float* __restrict__ Whh1,
    const float* __restrict__ bih1, const float* __restrict__ bhh1,
    const float* __restrict__ Wih2, const float* __restrict__ Whh2,
    const float* __restrict__ bih2, const float* __restrict__ bhh2,
    float* __restrict__ out)
{
  // Extended, gate-padded weight matrices in LDS (f16), K-slot permuted:
  //   W1_ext[m=256][klog=64]: m = gate*64 + row (row<51 valid);
  //     klog<51 -> Whh1[gm][klog], klog==51 -> Wih1[gm],
  //     klog==52 -> bih1[gm]+bhh1[gm], else 0.
  //   Gate rows i/f/o (gate!=2) scaled by 0.5 (folded sigmoid pre-scale).
  //   Stored at slot K where klog = swap34(K).
  __shared__ _Float16 w1s[256 * 64];
  __shared__ _Float16 w2s[16 * 64];

  for (int idx = threadIdx.x; idx < 256 * 64; idx += blockDim.x) {
    int m = idx >> 6, k = swap34(idx & 63);
    int gate = m >> 6, row = m & 63;
    float v = 0.0f;
    if (row < H1c) {
      int gm = gate * H1c + row;
      if (k < H1c)            v = Whh1[gm * H1c + k];
      else if (k == H1c)      v = Wih1[gm];
      else if (k == H1c + 1)  v = bih1[gm] + bhh1[gm];
      if (gate != 2) v *= 0.5f;           // fold sigmoid 0.5 pre-scale
    }
    w1s[idx] = (_Float16)v;
  }
  for (int idx = threadIdx.x; idx < 16 * 64; idx += blockDim.x) {
    int m = idx >> 6, k = swap34(idx & 63);
    float v = 0.0f;
    if (m < 4) {
      if (k < H1c)            v = Wih2[m * H1c + k];
      else if (k == H1c)      v = Whh2[m];
      else if (k == H1c + 1)  v = bih2[m] + bhh2[m];
      if (m != 2) v *= 0.5f;
    }
    w2s[idx] = (_Float16)v;
  }
  __syncthreads();

  const int lane = threadIdx.x & 31;
  const int hi   = lane >> 4;   // half-wave: 0 = lanes 0-15, 1 = lanes 16-31
  const int col  = lane & 15;   // batch column within the 16-wide tile
  const int wave = threadIdx.x >> 5;
  const int b    = (blockIdx.x * (blockDim.x >> 5) + wave) * 16 + col;

  // Resident A fragments (contiguous k = 16*hi + s per lane half) — two ds_load_b128 each.
  v16h a1[16][2];
  #pragma unroll
  for (int mt = 0; mt < 16; ++mt) {
    #pragma unroll
    for (int kt = 0; kt < 2; ++kt) {
      int base = (mt * 16 + col) * 64 + kt * 32 + hi * 16;
      v16h a;
      *((v8h*)&a)     = *(const v8h*)&w1s[base];
      *((v8h*)&a + 1) = *(const v8h*)&w1s[base + 8];
      a1[mt][kt] = a;
    }
  }
  v16h a2[2];
  #pragma unroll
  for (int kt = 0; kt < 2; ++kt) {
    int base = col * 64 + kt * 32 + hi * 16;
    v16h a;
    *((v8h*)&a)     = *(const v8h*)&w2s[base];
    *((v8h*)&a + 1) = *(const v8h*)&w2s[base + 8];
    a2[kt] = a;
  }

  // State: c1 in f32 D-layout (tile j, slot v <-> klog = 16j + 8*hi + v).
  // Padded entries (klog>=51) stay exactly 0 since their weight rows are 0.
  float c1s[4][8];
  #pragma unroll
  for (int j = 0; j < 4; ++j)
    #pragma unroll
    for (int v = 0; v < 8; ++v) c1s[j][v] = 0.0f;
  float c2 = 0.0f, h2 = 0.0f;

  const float* xrow = input + (long)b * Tc;
  float*       orow = out   + (long)b * Tc;

  // Initial B operand: h1 = 0; x_0 at logical k=51 -> slot 43 (kt=1, elem 11,
  // lanes 0-15); 1.0 at logical k=52 -> slot 44 (kt=1, elem 12, lanes 0-15).
  v16h b1k0 = zero16();
  v16h b1k1 = zero16();
  {
    float x0 = xrow[0];
    b1k1[11] = hi ? b1k1[11] : (_Float16)x0;
    b1k1[12] = hi ? b1k1[12] : (_Float16)1.0f;
  }

  #pragma unroll 1
  for (int t = 0; t < Tc; ++t) {
    float xn = xrow[(t + 1 < Tc) ? (t + 1) : t];  // prefetch next input

    // ---- layer 1: 32 WMMAs + fused elementwise per k-group j ----
    // With the swap34 K-permutation, D-tile j feeds B-slot dwords
    // [kt = j>>1].u[4*(j&1) + d] directly — no cross-lane movement.
    U16 NB[2], CB[2];
    #pragma unroll
    for (int j = 0; j < 4; ++j) {
      v8f z = {};
      v8f gi = WMMA(a1[j][0],      b1k0, z);
      gi     = WMMA(a1[j][1],      b1k1, gi);
      v8f gf = WMMA(a1[4 + j][0],  b1k0, z);
      gf     = WMMA(a1[4 + j][1],  b1k1, gf);
      v8f gg = WMMA(a1[8 + j][0],  b1k0, z);
      gg     = WMMA(a1[8 + j][1],  b1k1, gg);
      v8f go = WMMA(a1[12 + j][0], b1k0, z);
      go     = WMMA(a1[12 + j][1], b1k1, go);
      #pragma unroll
      for (int v = 0; v < 8; v += 2) {
        float ca = __builtin_fmaf(sigh(gf[v]),     c1s[j][v],
                                  sigh(gi[v])     * tanh1(gg[v]));
        float cb = __builtin_fmaf(sigh(gf[v + 1]), c1s[j][v + 1],
                                  sigh(gi[v + 1]) * tanh1(gg[v + 1]));
        c1s[j][v]     = ca;
        c1s[j][v + 1] = cb;
        float ha = sigh(go[v])     * tanh1(ca);
        float hb = sigh(go[v + 1]) * tanh1(cb);
        const int d = ((j & 1) << 2) + (v >> 1);
        NB[j >> 1].u[d] = pkf16(ha, hb);
        CB[j >> 1].u[d] = pkf16(ca, cb);
      }
    }
    v16h nb0 = NB[0].v, nb1 = NB[1].v, cb0 = CB[0].v, cb1 = CB[1].v;

    // ---- layer 2: c1_ext = [c1, h2 @slot43, 1.0 @slot44] (lo lanes) ----
    cb1[11] = hi ? cb1[11] : (_Float16)h2;   // h2 already lives on lo lanes
    cb1[12] = hi ? cb1[12] : (_Float16)1.0f;
    v8f z2 = {};
    v8f g2 = WMMA(a2[0], cb0, z2);
    g2     = WMMA(a2[1], cb1, g2);
    // lo lanes: g2[0..3] = (i2,f2,g2,o2) for column `col` (i/f/o pre-scaled)
    float c2n = __builtin_fmaf(sigh(g2[1]), c2, sigh(g2[0]) * tanh1(g2[2]));
    c2 = c2n;
    h2 = sigh(g2[3]) * tanh1(c2n);
    if (!hi) __builtin_nontemporal_store(c2n, &orow[t]);  // output is c2

    // insert x_{t+1} / 1.0 into next-step B and rotate
    nb1[11] = hi ? nb1[11] : (_Float16)xn;
    nb1[12] = hi ? nb1[12] : (_Float16)1.0f;
    b1k0 = nb0;
    b1k1 = nb1;
  }
}

extern "C" void kernel_launch(void* const* d_in, const int* in_sizes, int n_in,
                              void* d_out, int out_size, void* d_ws, size_t ws_size,
                              hipStream_t stream) {
  const float* input = (const float*)d_in[0];
  const float* Wih1  = (const float*)d_in[1];
  const float* Whh1  = (const float*)d_in[2];
  const float* bih1  = (const float*)d_in[3];
  const float* bhh1  = (const float*)d_in[4];
  const float* Wih2  = (const float*)d_in[5];
  const float* Whh2  = (const float*)d_in[6];
  const float* bih2  = (const float*)d_in[7];
  const float* bhh2  = (const float*)d_in[8];
  float* out = (float*)d_out;

  const int B = in_sizes[0] / Tc;       // 8192 for the reference harness
  const int blocks = B / 64;            // 4 waves/block * 16 batch cols/wave
  lstm2_wmma_kernel<<<dim3(blocks), dim3(128), 0, stream>>>(
      input, Wih1, Whh1, bih1, bhh1, Wih2, Whh2, bih2, bhh2, out);
}
